// PatternValueDualRetriever_7928509628660
// MI455X (gfx1250) — compile-verified, matrix operand order checked by reading, not run on hardware
//
#include <hip/hip_runtime.h>
#include <hip/hip_bf16.h>
#include <math.h>

// ---------------- problem constants ----------------
#define S_ 4
#define B_ 128
#define T_ 512
#define N_ 7
#define M_ 200000
#define D_ 128
#define P_ 96
#define K_ 16
#define SIGMA_ 3.0f
#define RED_ 0.2f

// ---------------- GEMM blocking ----------------
#define BROWS 64            // batch rows per block (4 x 16-row WMMA tiles)
#define BG 2                // B_/BROWS
#define BTILES 4            // 16-row tiles per block
#define CTILES 32           // 16-col m-tiles per chunk
#define CCOLS 512           // CTILES*16
#define NCHUNK 391          // ceil(12500 tiles / 32)

typedef __attribute__((ext_vector_type(16))) __bf16 v16bf;
typedef __attribute__((ext_vector_type(8)))  float  v8f;

// native f32 -> bf16 convert (lowered by the backend; avoids 3-op RTNE emulation)
__device__ __forceinline__ __bf16 f2bf(float f) { return (__bf16)f; }

// sorted-descending top-16 insert via swap chain (static indexing only)
__device__ __forceinline__ void topk_insert(float (&tv)[16], int (&ti)[16],
                                            float v, int gi) {
  if (!(v > tv[15])) return;
  float cv = v; int ci = gi;
#pragma unroll
  for (int i = 0; i < 16; ++i) {
    bool gt = cv > tv[i];
    float nv = gt ? cv : tv[i]; int ni = gt ? ci : ti[i];
    float ov = gt ? tv[i] : cv; int oi = gt ? ti[i] : ci;
    tv[i] = nv; ti[i] = ni; cv = ov; ci = oi;
  }
}

// ==================== kernel 1: extreme detector ====================
// has_extreme[b] = any_n( max_t x > mu+3sd  or  min_t x < mu-3sd ), std ddof=1
__global__ void k_extreme(const float* __restrict__ x, float* __restrict__ extf) {
  __shared__ float rs[256], rq[256], rmn[256], rmx[256];
  const int b = blockIdx.x, tid = threadIdx.x;
  int ex = 0;
  for (int n = 0; n < N_; ++n) {
    float s = 0.f, q = 0.f, mn = __builtin_inff(), mx = -__builtin_inff();
    for (int t = tid; t < T_; t += 256) {
      float v = x[((size_t)b * T_ + t) * N_ + n];
      s += v; q += v * v; mn = fminf(mn, v); mx = fmaxf(mx, v);
    }
    rs[tid] = s; rq[tid] = q; rmn[tid] = mn; rmx[tid] = mx;
    __syncthreads();
    for (int o = 128; o > 0; o >>= 1) {
      if (tid < o) {
        rs[tid] += rs[tid + o]; rq[tid] += rq[tid + o];
        rmn[tid] = fminf(rmn[tid], rmn[tid + o]);
        rmx[tid] = fmaxf(rmx[tid], rmx[tid + o]);
      }
      __syncthreads();
    }
    if (tid == 0) {
      float mean = rs[0] / (float)T_;
      float var = (rq[0] - (float)T_ * mean * mean) / (float)(T_ - 1);
      float sd = fmaxf(sqrtf(fmaxf(var, 0.f)), 1e-6f);
      if (rmx[0] - mean > SIGMA_ * sd || mean - rmn[0] > SIGMA_ * sd) ex = 1;
    }
    __syncthreads();
  }
  if (tid == 0) extf[b] = ex ? 1.f : 0.f;
}

// ==================== kernel 2: query encoder ====================
// per (s,b): downsample -> 5 stats -> Linear(5,D) -> LayerNorm -> L2 normalize
__global__ void k_encode(const float* __restrict__ x, const float* __restrict__ W,
                         const float* __restrict__ bv, const float* __restrict__ gamma,
                         const float* __restrict__ beta, float* __restrict__ qk) {
  __shared__ float red[128];
  __shared__ float stats[8];   // [0..4]=mean,std,max,min,trend  [5]=first [6]=last
  __shared__ float sc[4];
  const int gid = blockIdx.x;
  const int s = gid / B_, b = gid % B_;
  const int tid = threadIdx.x;
  const int r = 1 << s, Tds = T_ >> s, grp = r * N_;
  const float inv = 1.f / (float)grp;
  const float* xb = x + (size_t)b * T_ * N_;

  float sum = 0.f, sq = 0.f, mn = __builtin_inff(), mx = -__builtin_inff();
  for (int t = tid; t < Tds; t += 128) {
    const float* p = xb + (size_t)t * grp;
    float a = 0.f;
    for (int j = 0; j < grp; ++j) a += p[j];
    a *= inv;
    sum += a; sq += a * a; mn = fminf(mn, a); mx = fmaxf(mx, a);
    if (t == 0) stats[5] = a;
    if (t == Tds - 1) stats[6] = a;
  }
  // four block reductions
  red[tid] = sum; __syncthreads();
  for (int o = 64; o > 0; o >>= 1) { if (tid < o) red[tid] += red[tid + o]; __syncthreads(); }
  if (tid == 0) sc[0] = red[0];
  __syncthreads();
  red[tid] = sq; __syncthreads();
  for (int o = 64; o > 0; o >>= 1) { if (tid < o) red[tid] += red[tid + o]; __syncthreads(); }
  if (tid == 0) sc[1] = red[0];
  __syncthreads();
  red[tid] = mn; __syncthreads();
  for (int o = 64; o > 0; o >>= 1) { if (tid < o) red[tid] = fminf(red[tid], red[tid + o]); __syncthreads(); }
  if (tid == 0) sc[2] = red[0];
  __syncthreads();
  red[tid] = mx; __syncthreads();
  for (int o = 64; o > 0; o >>= 1) { if (tid < o) red[tid] = fmaxf(red[tid], red[tid + o]); __syncthreads(); }
  if (tid == 0) {
    float mean = sc[0] / (float)Tds;
    float var = (sc[1] - (float)Tds * mean * mean) / (float)(Tds - 1);
    float sd = fmaxf(sqrtf(fmaxf(var, 0.f)), 1e-6f);
    stats[0] = mean; stats[1] = sd; stats[2] = red[0]; stats[3] = sc[2];
    stats[4] = stats[6] - stats[5];
  }
  __syncthreads();

  // h = stats @ W.T + b ; one d per thread
  const int d = tid;
  float h = bv[d];
#pragma unroll
  for (int j = 0; j < 5; ++j) h += stats[j] * W[d * 5 + j];

  // LayerNorm over D (biased var, eps 1e-5)
  red[tid] = h; __syncthreads();
  for (int o = 64; o > 0; o >>= 1) { if (tid < o) red[tid] += red[tid + o]; __syncthreads(); }
  if (tid == 0) sc[0] = red[0] / (float)D_;
  __syncthreads();
  float mu = sc[0];
  red[tid] = (h - mu) * (h - mu); __syncthreads();
  for (int o = 64; o > 0; o >>= 1) { if (tid < o) red[tid] += red[tid + o]; __syncthreads(); }
  if (tid == 0) sc[1] = red[0] / (float)D_;
  __syncthreads();
  float g = (h - mu) * rsqrtf(sc[1] + 1e-5f) * gamma[d] + beta[d];

  // L2 normalize over D
  red[tid] = g * g; __syncthreads();
  for (int o = 64; o > 0; o >>= 1) { if (tid < o) red[tid] += red[tid + o]; __syncthreads(); }
  if (tid == 0) sc[2] = fmaxf(sqrtf(red[0]), 1e-12f);
  __syncthreads();
  qk[((size_t)(s * B_) + b) * D_ + d] = g / sc[2];
}

// ==================== kernel 3: bf16 WMMA similarity + per-chunk top-16 ====================
// grid = S_*BG*NCHUNK, block = 256 (8 waves). LDS: sim[64][512] f32 + 256 lists.
__global__ void __launch_bounds__(256)
k_sim_topk(const float* __restrict__ qk, const float* __restrict__ mem_keys,
           float* __restrict__ cval, int* __restrict__ cidx) {
  extern __shared__ float sm[];
  float* sim  = sm;                              // BROWS*CCOLS floats
  float* lval = sm + BROWS * CCOLS;              // 256*16 floats
  int*   lidx = (int*)(lval + 256 * 16);         // 256*16 ints

  const int gid = blockIdx.x;
  const int chunk = gid % NCHUNK;
  const int bg = (gid / NCHUNK) % BG;
  const int s = gid / (NCHUNK * BG);
  const int tid = threadIdx.x;
  const int wave = tid >> 5, lane = tid & 31;
  const int l16 = lane & 15;
  const int halfsel = (lane >> 4) ? 8 : 0;       // bf16 A/B lane K-group select

  // ---- preload A operands (queries): A[bt][kstep], 16x32 bf16 tile each ----
  v16bf A[BTILES][4];
#pragma unroll
  for (int bt = 0; bt < BTILES; ++bt) {
    const float* qrow = qk + ((size_t)(s * B_) + bg * BROWS + bt * 16 + l16) * D_;
#pragma unroll
    for (int kb = 0; kb < 4; ++kb) {
      const int k0 = kb * 32 + halfsel;
      float4 p0 = *(const float4*)(qrow + k0);
      float4 p1 = *(const float4*)(qrow + k0 + 4);
      float4 p2 = *(const float4*)(qrow + k0 + 16);
      float4 p3 = *(const float4*)(qrow + k0 + 20);
      A[bt][kb][0] = f2bf(p0.x); A[bt][kb][1] = f2bf(p0.y);
      A[bt][kb][2] = f2bf(p0.z); A[bt][kb][3] = f2bf(p0.w);
      A[bt][kb][4] = f2bf(p1.x); A[bt][kb][5] = f2bf(p1.y);
      A[bt][kb][6] = f2bf(p1.z); A[bt][kb][7] = f2bf(p1.w);
      A[bt][kb][8]  = f2bf(p2.x); A[bt][kb][9]  = f2bf(p2.y);
      A[bt][kb][10] = f2bf(p2.z); A[bt][kb][11] = f2bf(p2.w);
      A[bt][kb][12] = f2bf(p3.x); A[bt][kb][13] = f2bf(p3.y);
      A[bt][kb][14] = f2bf(p3.z); A[bt][kb][15] = f2bf(p3.w);
    }
  }

  // ---- GEMM phase: each wave handles 4 m-tiles; B tile reused over 4 batch tiles ----
  const float4 z4 = make_float4(0.f, 0.f, 0.f, 0.f);
  for (int tt = 0; tt < 4; ++tt) {
    const int tloc = wave * 4 + tt;
    const int mtile = chunk * CTILES + tloc;
    const long mcol = (long)mtile * 16 + l16;
    const bool ok = (mcol < (long)M_);
    const float* krow = mem_keys + ((size_t)s * M_ + (ok ? (size_t)mcol : 0)) * D_;
    __builtin_prefetch(krow + 16 * D_, 0, 1);    // next m-tile for this wave: +16 key rows
    v16bf Bv[4];
#pragma unroll
    for (int kb = 0; kb < 4; ++kb) {
      const int k0 = kb * 32 + halfsel;
      float4 p0 = ok ? *(const float4*)(krow + k0)      : z4;
      float4 p1 = ok ? *(const float4*)(krow + k0 + 4)  : z4;
      float4 p2 = ok ? *(const float4*)(krow + k0 + 16) : z4;
      float4 p3 = ok ? *(const float4*)(krow + k0 + 20) : z4;
      Bv[kb][0] = f2bf(p0.x); Bv[kb][1] = f2bf(p0.y);
      Bv[kb][2] = f2bf(p0.z); Bv[kb][3] = f2bf(p0.w);
      Bv[kb][4] = f2bf(p1.x); Bv[kb][5] = f2bf(p1.y);
      Bv[kb][6] = f2bf(p1.z); Bv[kb][7] = f2bf(p1.w);
      Bv[kb][8]  = f2bf(p2.x); Bv[kb][9]  = f2bf(p2.y);
      Bv[kb][10] = f2bf(p2.z); Bv[kb][11] = f2bf(p2.w);
      Bv[kb][12] = f2bf(p3.x); Bv[kb][13] = f2bf(p3.y);
      Bv[kb][14] = f2bf(p3.z); Bv[kb][15] = f2bf(p3.w);
    }
#pragma unroll
    for (int bt = 0; bt < BTILES; ++bt) {
      v8f acc = {0.f, 0.f, 0.f, 0.f, 0.f, 0.f, 0.f, 0.f};
#pragma unroll
      for (int kb = 0; kb < 4; ++kb)
        acc = __builtin_amdgcn_wmma_f32_16x16x32_bf16(
            false, A[bt][kb], false, Bv[kb], (short)0, acc, false, false);
      const int colbase = tloc * 16 + l16;
      const int rbase = bt * 16 + ((lane >> 4) << 3);
#pragma unroll
      for (int j = 0; j < 8; ++j)
        sim[(rbase + j) * CCOLS + colbase] = acc[j];
    }
  }
  __syncthreads();

  // ---- per-row top-16 over this chunk: 4 threads per row ----
  float tv[16]; int ti[16];
#pragma unroll
  for (int i = 0; i < 16; ++i) { tv[i] = -__builtin_inff(); ti[i] = 0; }
  const int row = tid >> 2, sub = tid & 3;
  const long gbase = (long)chunk * CCOLS;
  for (int c = sub; c < CCOLS; c += 4) {
    const long g = gbase + c;
    if (g >= (long)M_) break;
    topk_insert(tv, ti, sim[row * CCOLS + c], (int)g);
  }
#pragma unroll
  for (int i = 0; i < 16; ++i) { lval[tid * 16 + i] = tv[i]; lidx[tid * 16 + i] = ti[i]; }
  __syncthreads();

  if (tid < BROWS) {
    float fv[16]; int fi[16];
#pragma unroll
    for (int i = 0; i < 16; ++i) { fv[i] = -__builtin_inff(); fi[i] = 0; }
    const int base = tid * 4;
    for (int k = 0; k < 4; ++k)
      for (int i = 0; i < 16; ++i)
        topk_insert(fv, fi, lval[(base + k) * 16 + i], lidx[(base + k) * 16 + i]);
    const size_t cb = ((size_t)((s * BG + bg) * NCHUNK + chunk)) * (BROWS * 16) + (size_t)tid * 16;
    for (int i = 0; i < 16; ++i) { cval[cb + i] = fv[i]; cidx[cb + i] = fi[i]; }
  }
}

// ==================== kernel 4: candidate merge + softmax + gather + broadcast ====================
__global__ void k_final(const float* __restrict__ cval, const int* __restrict__ cidx,
                        const float* __restrict__ thresholds, const float* __restrict__ extf,
                        const float* __restrict__ mem_values, float* __restrict__ out) {
  __shared__ float lv[128 * 16]; __shared__ int li[128 * 16];
  __shared__ float l2v[16 * 16]; __shared__ int l2i[16 * 16];
  __shared__ float wsh[16]; __shared__ int ish[16]; __shared__ float validf;
  const int gid = blockIdx.x;
  const int s = gid / B_, b = gid % B_;
  const int bg = b / BROWS, row = b % BROWS;
  const int tid = threadIdx.x;
  const int NC = NCHUNK * 16;

  float tv[16]; int ti[16];
#pragma unroll
  for (int i = 0; i < 16; ++i) { tv[i] = -__builtin_inff(); ti[i] = 0; }
  for (int c = tid; c < NC; c += 128) {
    const int chunk = c >> 4, i = c & 15;
    const size_t a = ((size_t)((s * BG + bg) * NCHUNK + chunk)) * (BROWS * 16) + (size_t)row * 16 + i;
    topk_insert(tv, ti, cval[a], cidx[a]);
  }
#pragma unroll
  for (int i = 0; i < 16; ++i) { lv[tid * 16 + i] = tv[i]; li[tid * 16 + i] = ti[i]; }
  __syncthreads();

  if (tid < 16) {
    float fv[16]; int fi[16];
#pragma unroll
    for (int i = 0; i < 16; ++i) { fv[i] = -__builtin_inff(); fi[i] = 0; }
    for (int k = 0; k < 8; ++k)
      for (int i = 0; i < 16; ++i)
        topk_insert(fv, fi, lv[(tid * 8 + k) * 16 + i], li[(tid * 8 + k) * 16 + i]);
    for (int i = 0; i < 16; ++i) { l2v[tid * 16 + i] = fv[i]; l2i[tid * 16 + i] = fi[i]; }
  }
  __syncthreads();

  if (tid == 0) {
    float fv[16]; int fi[16];
#pragma unroll
    for (int i = 0; i < 16; ++i) { fv[i] = -__builtin_inff(); fi[i] = 0; }
    for (int k = 0; k < 256; ++k) topk_insert(fv, fi, l2v[k], l2i[k]);
    float thr = 1.f / (1.f + expf(-thresholds[s])) - extf[b] * RED_;
    thr = fmaxf(thr, 0.1f);
    const float vmax = fv[0];
    validf = (vmax > thr) ? 1.f : 0.f;
    float e[16]; float ssum = 0.f;
#pragma unroll
    for (int i = 0; i < 16; ++i) { e[i] = expf(fv[i] - vmax); ssum += e[i]; }
#pragma unroll
    for (int i = 0; i < 16; ++i) { wsh[i] = e[i] / ssum; ish[i] = fi[i]; }
  }
  __syncthreads();

  for (int p = tid; p < P_; p += 128) {
    float h = 0.f;
#pragma unroll
    for (int k = 0; k < 16; ++k)
      h += wsh[k] * mem_values[((size_t)s * M_ + (size_t)ish[k]) * P_ + p];
    h *= validf;
    float* ob = out + ((size_t)(s * B_ + b) * P_ + p) * N_;
#pragma unroll
    for (int n = 0; n < N_; ++n) ob[n] = h;
  }
}

// ==================== launch ====================
extern "C" void kernel_launch(void* const* d_in, const int* in_sizes, int n_in,
                              void* d_out, int out_size, void* d_ws, size_t ws_size,
                              hipStream_t stream) {
  (void)in_sizes; (void)n_in; (void)out_size; (void)ws_size;
  const float* x          = (const float*)d_in[0];
  const float* W          = (const float*)d_in[1];
  const float* bv         = (const float*)d_in[2];
  const float* gamma      = (const float*)d_in[3];
  const float* beta       = (const float*)d_in[4];
  const float* thresholds = (const float*)d_in[5];
  const float* mem_keys   = (const float*)d_in[6];
  const float* mem_values = (const float*)d_in[7];
  float* out = (float*)d_out;

  char* ws = (char*)d_ws;
  float* qk   = (float*)ws;                                  // S*B*D f32
  float* extf = (float*)(ws + (size_t)S_ * B_ * D_ * 4);     // B f32
  size_t off = (size_t)S_ * B_ * D_ * 4 + 1024;
  const size_t ncand = (size_t)S_ * BG * NCHUNK * BROWS * 16;
  float* cval = (float*)(ws + off);
  int*   cidx = (int*)(ws + off + ncand * 4);

  k_extreme<<<B_, 256, 0, stream>>>(x, extf);
  k_encode<<<S_ * B_, 128, 0, stream>>>(x, W, bv, gamma, beta, qk);

  const size_t smem = (size_t)BROWS * CCOLS * 4 + 256 * 16 * 4 + 256 * 16 * 4;
  k_sim_topk<<<S_ * BG * NCHUNK, 256, smem, stream>>>(qk, mem_keys, cval, cidx);

  k_final<<<S_ * B_, 128, 0, stream>>>(cval, cidx, thresholds, extf, mem_values, out);
}
